// TokenRoutedMLP_87016037417298
// MI455X (gfx1250) — matrix operand order repaired and checked by reading, not compiled, not run on hardware
//
#include <hip/hip_runtime.h>
#include <hip/hip_bf16.h>

#define HIDDEN 2048
#define NE 8
#define EI 2048
#define NT 4096

typedef __attribute__((ext_vector_type(16))) __bf16          v16bf;
typedef __attribute__((ext_vector_type(16))) unsigned short  v16u;
typedef __attribute__((ext_vector_type(8)))  float           v8f;

__device__ __forceinline__ unsigned short f2bf(float f) {
  unsigned int u = __float_as_uint(f);
  unsigned int r = u + 0x7FFFu + ((u >> 16) & 1u);   // round-to-nearest-even
  return (unsigned short)(r >> 16);
}

// packed fp32x2 -> bf16x2 (v_cvt_pk_bf16_f32 when available)
__device__ __forceinline__ unsigned int pk2bf(float a, float b) {
#if __has_builtin(__builtin_amdgcn_cvt_pk_bf16_f32)
  typedef __attribute__((ext_vector_type(2))) __bf16 v2bf;
  v2bf r = __builtin_amdgcn_cvt_pk_bf16_f32(a, b);
  return __builtin_bit_cast(unsigned int, r);
#else
  return (unsigned int)f2bf(a) | ((unsigned int)f2bf(b) << 16);
#endif
}

// per-lane async copy of 16B global -> LDS (tracked by ASYNCcnt)
__device__ __forceinline__ void async_copy_b128(unsigned lds_addr, const void* gaddr) {
  asm volatile("global_load_async_to_lds_b128 %0, %1, off"
               :: "v"(lds_addr), "v"((unsigned long long)gaddr)
               : "memory");
}
__device__ __forceinline__ void wait_async() {
  asm volatile("s_wait_asynccnt 0x0" ::: "memory");
}

// ---------------- routing: bucket tokens by expert ----------------
__global__ void k_route(const int* __restrict__ tok_ids,
                        int* __restrict__ counts,
                        int* __restrict__ tlist) {
  int t = blockIdx.x * blockDim.x + threadIdx.x;
  if (t < NT) {
    int e = tok_ids[t] & (NE - 1);                   // ids are non-negative
    int slot = atomicAdd(&counts[e], 1);
    tlist[e * NT + slot] = t;
  }
}

// ---------------- convert x fp32 -> bf16 --------------------------
__global__ void k_cvt(const float* __restrict__ x, unsigned short* __restrict__ xb) {
  int i = (blockIdx.x * 256 + threadIdx.x) * 4;
  float4 v = *(const float4*)(x + i);
  uint2 r;
  r.x = pk2bf(v.x, v.y);
  r.y = pk2bf(v.z, v.w);
  *(uint2*)(xb + i) = r;
}

// ---------------- gate_up GEMM + silu*up --------------------------
// grid: (NT/64, EI/128, NE), block: 256 (8 waves)
// block tile: 64(M) x 128(N) for gate AND up; wave: 16(M) x 64(N) of each
__global__ __launch_bounds__(256)
void k_gateup(const unsigned short* __restrict__ xb,
              const float* __restrict__ gup,
              const int* __restrict__ counts,
              const int* __restrict__ tlist,
              unsigned short* __restrict__ hb) {
  const int e   = blockIdx.z;
  const int cnt = counts[e];
  const int m0  = blockIdx.x * 64;
  if (m0 >= cnt) return;
  const int n0  = blockIdx.y * 128;

  __shared__ unsigned short As[64][40];    // [m][k] bf16, padded
  __shared__ unsigned short Bg[128][40];   // [n][k] bf16 (gate)
  __shared__ unsigned short Bu[128][40];   // [n][k] bf16 (up)
  __shared__ int rows[64];

  const int tIdx = threadIdx.x;
  if (tIdx < 64) {
    int m = m0 + tIdx;
    rows[tIdx] = (m < cnt) ? tlist[e * NT + m] : -1;
  }
  __syncthreads();

  // loader coordinates
  const int a_r = tIdx >> 2;             // 0..63 (m)
  const int a_c = (tIdx & 3) * 8;        // 0,8,16,24 (k)
  const int b_r = tIdx >> 3;             // 0..31 (k)
  const int b_c = (tIdx & 7) * 16;       // 0..112 (n)

  // wave coordinates
  const int wave   = tIdx >> 5;
  const int lane   = tIdx & 31;
  const int lhalf  = lane >> 4;
  const int lrow   = lane & 15;
  const int mstrip = (wave & 3) * 16;    // 16-row strip
  const int nquad  = (wave >> 2) * 64;   // 64-col half

  v8f ag[4] = {v8f{}, v8f{}, v8f{}, v8f{}};
  v8f au[4] = {v8f{}, v8f{}, v8f{}, v8f{}};

  int tokA = rows[a_r]; if (tokA < 0) tokA = rows[0];
  const unsigned short* xrow = xb + (size_t)tokA * HIDDEN + a_c;
  const unsigned a_lds = (unsigned)(unsigned long long)&As[a_r][a_c];
  const size_t wbase = (size_t)e * HIDDEN * (2 * EI);
  const float* bsrc0 = gup + wbase + (size_t)b_r * (2 * EI) + n0 + b_c;

  for (int k0 = 0; k0 < HIDDEN; k0 += 32) {
    __syncthreads();
    // A tile: async global->LDS, 16B per lane
    async_copy_b128(a_lds, xrow + k0);
    // B tiles: fp32 -> bf16, transpose into [n][k]
    const float* gsrc = bsrc0 + (size_t)k0 * (2 * EI);
    const float* usrc = gsrc + EI;
    #pragma unroll
    for (int q = 0; q < 4; ++q) {
      float4 g = *(const float4*)(gsrc + q * 4);
      float4 u = *(const float4*)(usrc + q * 4);
      unsigned pg0 = pk2bf(g.x, g.y), pg1 = pk2bf(g.z, g.w);
      unsigned pu0 = pk2bf(u.x, u.y), pu1 = pk2bf(u.z, u.w);
      int c = b_c + q * 4;
      Bg[c + 0][b_r] = (unsigned short)pg0; Bg[c + 1][b_r] = (unsigned short)(pg0 >> 16);
      Bg[c + 2][b_r] = (unsigned short)pg1; Bg[c + 3][b_r] = (unsigned short)(pg1 >> 16);
      Bu[c + 0][b_r] = (unsigned short)pu0; Bu[c + 1][b_r] = (unsigned short)(pu0 >> 16);
      Bu[c + 2][b_r] = (unsigned short)pu1; Bu[c + 3][b_r] = (unsigned short)(pu1 >> 16);
    }
    wait_async();
    __syncthreads();

    if (k0 + 32 < HIDDEN) __builtin_prefetch(gsrc + (size_t)32 * (2 * EI), 0, 1);

    // ---- load ALL fragments first (lets backend clause the ds_loads), then WMMA chain ----
    // A fragment (16x32): lanes<16 hold K 0-15, lanes>=16 hold K 16-31
    v16u a;
    const unsigned short* Arow = &As[mstrip + lrow][0];
    const int ka0 = lhalf * 16, ka1 = lhalf * 16 + 8;
    #pragma unroll
    for (int i = 0; i < 8; ++i) { a[i] = Arow[ka0 + i]; a[8 + i] = Arow[ka1 + i]; }
    v16bf av = __builtin_bit_cast(v16bf, a);

    // B fragments (32x16): lanes<16 hold K 0-15, lanes>=16 hold K 16-31
    const int kb = lhalf * 16;
    v16u bg[4], bu[4];
    #pragma unroll
    for (int j = 0; j < 4; ++j) {
      const unsigned short* Bgp = &Bg[nquad + j * 16 + lrow][0];
      const unsigned short* Bup = &Bu[nquad + j * 16 + lrow][0];
      #pragma unroll
      for (int i = 0; i < 16; ++i) { bg[j][i] = Bgp[kb + i]; bu[j][i] = Bup[kb + i]; }
    }
    #pragma unroll
    for (int j = 0; j < 4; ++j)
      ag[j] = __builtin_amdgcn_wmma_f32_16x16x32_bf16(false, av, false, __builtin_bit_cast(v16bf, bg[j]), (short)0, ag[j], false, false);
    #pragma unroll
    for (int j = 0; j < 4; ++j)
      au[j] = __builtin_amdgcn_wmma_f32_16x16x32_bf16(false, av, false, __builtin_bit_cast(v16bf, bu[j]), (short)0, au[j], false, false);
  }

  // epilogue: h = silu(g)*u, scatter by token (C/D layout: VGPR v -> M=v+8*lhalf, N=lrow)
  #pragma unroll
  for (int v = 0; v < 8; ++v) {
    int lm  = mstrip + v + lhalf * 8;
    int tok = rows[lm];
    if (tok >= 0) {
      size_t base = (size_t)tok * EI + n0 + nquad + lrow;
      #pragma unroll
      for (int j = 0; j < 4; ++j) {
        float g = ag[j][v], u = au[j][v];
        hb[base + j * 16] = f2bf((g / (1.0f + __expf(-g))) * u);
      }
    }
  }
}

// ---------------- down GEMM -> d_out ------------------------------
// grid: (NT/64, HIDDEN/128, NE), block: 256 (8 waves)
__global__ __launch_bounds__(256)
void k_down(const unsigned short* __restrict__ hb,
            const float* __restrict__ dwn,
            const int* __restrict__ counts,
            const int* __restrict__ tlist,
            float* __restrict__ out) {
  const int e   = blockIdx.z;
  const int cnt = counts[e];
  const int m0  = blockIdx.x * 64;
  if (m0 >= cnt) return;
  const int n0  = blockIdx.y * 128;

  __shared__ unsigned short As[64][40];    // [m][k]
  __shared__ unsigned short Bs[128][40];   // [n][k]
  __shared__ int rows[64];

  const int tIdx = threadIdx.x;
  if (tIdx < 64) {
    int m = m0 + tIdx;
    rows[tIdx] = (m < cnt) ? tlist[e * NT + m] : -1;
  }
  __syncthreads();

  const int a_r = tIdx >> 2;
  const int a_c = (tIdx & 3) * 8;
  const int b_r = tIdx >> 3;             // 0..31 (k)
  const int b_c = (tIdx & 7) * 16;       // 0..112 (n)

  const int wave   = tIdx >> 5;
  const int lane   = tIdx & 31;
  const int lhalf  = lane >> 4;
  const int lrow   = lane & 15;
  const int mstrip = (wave & 3) * 16;
  const int nquad  = (wave >> 2) * 64;

  v8f acc[4] = {v8f{}, v8f{}, v8f{}, v8f{}};

  int tokA = rows[a_r]; if (tokA < 0) tokA = rows[0];
  const unsigned short* hrow = hb + (size_t)tokA * EI + a_c;
  const unsigned a_lds = (unsigned)(unsigned long long)&As[a_r][a_c];
  const size_t wbase = (size_t)e * EI * HIDDEN;
  const float* bsrc0 = dwn + wbase + (size_t)b_r * HIDDEN + n0 + b_c;

  for (int k0 = 0; k0 < EI; k0 += 32) {
    __syncthreads();
    async_copy_b128(a_lds, hrow + k0);
    const float* dsrc = bsrc0 + (size_t)k0 * HIDDEN;
    #pragma unroll
    for (int q = 0; q < 4; ++q) {
      float4 d = *(const float4*)(dsrc + q * 4);
      unsigned p0 = pk2bf(d.x, d.y), p1 = pk2bf(d.z, d.w);
      int c = b_c + q * 4;
      Bs[c + 0][b_r] = (unsigned short)p0; Bs[c + 1][b_r] = (unsigned short)(p0 >> 16);
      Bs[c + 2][b_r] = (unsigned short)p1; Bs[c + 3][b_r] = (unsigned short)(p1 >> 16);
    }
    wait_async();
    __syncthreads();

    if (k0 + 32 < EI) __builtin_prefetch(dsrc + (size_t)32 * HIDDEN, 0, 1);

    // ---- all fragment loads first, then WMMA chain ----
    v16u a;
    const unsigned short* Arow = &As[mstrip + lrow][0];
    const int ka0 = lhalf * 16, ka1 = lhalf * 16 + 8;
    #pragma unroll
    for (int i = 0; i < 8; ++i) { a[i] = Arow[ka0 + i]; a[8 + i] = Arow[ka1 + i]; }
    v16bf av = __builtin_bit_cast(v16bf, a);

    const int kb = lhalf * 16;
    v16u b[4];
    #pragma unroll
    for (int j = 0; j < 4; ++j) {
      const unsigned short* Bp = &Bs[nquad + j * 16 + lrow][0];
      #pragma unroll
      for (int i = 0; i < 16; ++i) { b[j][i] = Bp[kb + i]; }
    }
    #pragma unroll
    for (int j = 0; j < 4; ++j)
      acc[j] = __builtin_amdgcn_wmma_f32_16x16x32_bf16(false, av, false, __builtin_bit_cast(v16bf, b[j]), (short)0, acc[j], false, false);
  }

  #pragma unroll
  for (int v = 0; v < 8; ++v) {
    int lm  = mstrip + v + lhalf * 8;
    int tok = rows[lm];
    if (tok >= 0) {
      size_t base = (size_t)tok * HIDDEN + n0 + nquad + lrow;
      #pragma unroll
      for (int j = 0; j < 4; ++j) out[base + j * 16] = acc[j][v];
    }
  }
}

// ---------------- launcher ----------------------------------------
extern "C" void kernel_launch(void* const* d_in, const int* in_sizes, int n_in,
                              void* d_out, int out_size, void* d_ws, size_t ws_size,
                              hipStream_t stream) {
  const float* x       = (const float*)d_in[0];
  const int*   tok_ids = (const int*)d_in[1];
  const float* gup     = (const float*)d_in[2];
  const float* dwn     = (const float*)d_in[3];
  float*       out     = (float*)d_out;

  char* ws = (char*)d_ws;
  int*            counts = (int*)ws;                               // 256 B
  int*            tlist  = (int*)(ws + 256);                       // 128 KB
  unsigned short* xb     = (unsigned short*)(ws + 131328);         // 16 MB bf16 x
  unsigned short* hb     = (unsigned short*)(ws + 16908544);       // 16 MB bf16 h

  hipMemsetAsync(counts, 0, 256, stream);
  k_route<<<NT / 256, 256, 0, stream>>>(tok_ids, counts, tlist);
  k_cvt<<<(NT * HIDDEN) / 1024, 256, 0, stream>>>(x, xb);
  k_gateup<<<dim3(NT / 64, EI / 128, NE), 256, 0, stream>>>(xb, gup, counts, tlist, hb);
  k_down<<<dim3(NT / 64, HIDDEN / 128, NE), 256, 0, stream>>>(hb, dwn, counts, tlist, out);
}
